// GATEncoder_73340861546960
// MI455X (gfx1250) — compile-verified, matrix operand order checked by reading, not run on hardware
//
#include <hip/hip_runtime.h>
#include <hip/hip_bf16.h>

typedef __bf16 v8bf  __attribute__((ext_vector_type(8)));
typedef __bf16 v16bf __attribute__((ext_vector_type(16)));
typedef float  v8f   __attribute__((ext_vector_type(8)));

#define FDIM  512   // HEADS * CDIM for all GAT layers
#define NHEAD 4
#define CDIM  128

__device__ __forceinline__ float lrelu(float v, float s) { return v > 0.f ? v : s * v; }

__device__ __forceinline__ void atomicMaxF(float* addr, float val) {
  unsigned int* ua = (unsigned int*)addr;
  unsigned int old = *ua;
  while (__uint_as_float(old) < val) {
    unsigned int assumed = old;
    old = atomicCAS(ua, assumed, __float_as_uint(val));
    if (old == assumed) break;
  }
}

// ---------------------------------------------------------------------------
// f32 -> bf16 elementwise copy (prepares WMMA A operand)
// ---------------------------------------------------------------------------
__global__ void cvt_bf16(const float* __restrict__ src, __bf16* __restrict__ dst, long long n) {
  long long i = (long long)blockIdx.x * blockDim.x + threadIdx.x;
  if (i < n) dst[i] = (__bf16)src[i];
}

// f32 W[K,N] -> bf16 Wt[N,K] (transpose; prepares WMMA B operand so a
// column of B is a contiguous 16B-aligned run)
__global__ void cvt_tr_bf16(const float* __restrict__ W, __bf16* __restrict__ Wt,
                            int K, int N) {
  int idx = blockIdx.x * blockDim.x + threadIdx.x;
  if (idx >= K * N) return;
  const int k = idx / N, n = idx - k * N;
  Wt[(size_t)n * K + k] = (__bf16)W[idx];
}

// ---------------------------------------------------------------------------
// GEMM: C[M,N] = A[M,K] * B[K,N], A given as bf16 row-major, B given as
// bf16 transposed (Bt[N,K] row-major). f32 accumulate via WMMA 16x16x32.
// 8 waves/block; each wave computes a 16(M) x 64(N) strip (4 accumulators,
// shared A fragment). Fragments are two contiguous global_load_b128 per lane
// following the documented CDNA5 16-bit A/B VGPR layout.
// Requires: M % 16 == 0, N % 64 == 0, K % 32 == 0.
// ---------------------------------------------------------------------------
__global__ __launch_bounds__(256) void gemm_wmma_bf16(
    const __bf16* __restrict__ A, const __bf16* __restrict__ Bt,
    float* __restrict__ C, int M, int N, int K)
{
  const int lane = threadIdx.x & 31;
  const int wave = threadIdx.x >> 5;
  const int hl   = lane >> 4;      // half-lane group (K-striping select)
  const int r    = lane & 15;      // A row / B col within tile
  const int mt   = blockIdx.y * 8 + wave;
  if (mt * 16 >= M) return;
  const int m0 = mt * 16;
  const int n0 = blockIdx.x * 64;

  v8f acc[4] = {};
  const __bf16* Arow = A  + (size_t)(m0 + r) * K + hl * 8;
  const __bf16* Brow = Bt + (size_t)(n0 + r) * K + hl * 8;

  for (int kk = 0; kk < K; kk += 32) {
    // A fragment (16x32): lane = row; K runs [hl*8, hl*8+8) and [16+hl*8, ...)
    const v8bf alo = *(const v8bf*)(Arow + kk);
    const v8bf ahi = *(const v8bf*)(Arow + kk + 16);
    const v16bf a = __builtin_shufflevector(alo, ahi,
        0, 1, 2, 3, 4, 5, 6, 7, 8, 9, 10, 11, 12, 13, 14, 15);
    if (kk + 32 < K) {
      __builtin_prefetch(Arow + kk + 32, 0, 1);   // global_prefetch_b8
      __builtin_prefetch(Brow + kk + 32, 0, 1);
    }
#pragma unroll
    for (int t = 0; t < 4; ++t) {
      const __bf16* Br = Brow + (size_t)(t * 16) * K;
      const v8bf blo = *(const v8bf*)(Br + kk);
      const v8bf bhi = *(const v8bf*)(Br + kk + 16);
      const v16bf b = __builtin_shufflevector(blo, bhi,
          0, 1, 2, 3, 4, 5, 6, 7, 8, 9, 10, 11, 12, 13, 14, 15);
      acc[t] = __builtin_amdgcn_wmma_f32_16x16x32_bf16(
          false, a, false, b, (short)0, acc[t], false, false);
    }
  }

  // C/D layout: VGPR rr -> row m0 + rr + 8*hl, col = n0 + t*16 + (lane&15)
#pragma unroll
  for (int t = 0; t < 4; ++t) {
    const int col = n0 + t * 16 + r;
    float* Cp = C + (size_t)(m0 + hl * 8) * N + col;
#pragma unroll
    for (int rr = 0; rr < 8; ++rr) Cp[(size_t)rr * N] = acc[t][rr];
  }
}

// ---------------------------------------------------------------------------
__global__ void fill_f32(float* __restrict__ p, float v, long long n) {
  long long i = (long long)blockIdx.x * blockDim.x + threadIdx.x;
  if (i < n) p[i] = v;
}

// a_s[n,h] = sum_c h[n,h,c]*attS[h,c]; a_d likewise
__global__ void attn_coeff(const float* __restrict__ h, const float* __restrict__ attS,
                           const float* __restrict__ attD, float* __restrict__ as_,
                           float* __restrict__ ad_, int nh) {
  int idx = blockIdx.x * blockDim.x + threadIdx.x;
  if (idx >= nh) return;
  const int hd = idx & 3;
  const float* hp = h + (size_t)(idx >> 2) * FDIM + hd * CDIM;
  const float* sp = attS + hd * CDIM;
  const float* dp = attD + hd * CDIM;
  float ss = 0.f, sd = 0.f;
  for (int c = 0; c < CDIM; ++c) { const float v = hp[c]; ss += v * sp[c]; sd += v * dp[c]; }
  as_[idx] = ss; ad_[idx] = sd;
}

__global__ void edge_max(const long long* __restrict__ ei, long long E, long long ET,
                         const float* __restrict__ as_, const float* __restrict__ ad_,
                         float* __restrict__ mmax) {
  long long idx = (long long)blockIdx.x * blockDim.x + threadIdx.x;
  if (idx >= ET * NHEAD) return;
  const long long e = idx >> 2; const int hd = (int)(idx & 3);
  long long s, d;
  if (e < E) { s = ei[e]; d = ei[E + e]; } else { s = d = e - E; }
  const float sc = lrelu(as_[s * NHEAD + hd] + ad_[d * NHEAD + hd], 0.2f);
  atomicMaxF(&mmax[d * NHEAD + hd], sc);
}

__global__ void edge_exp(const long long* __restrict__ ei, long long E, long long ET,
                         const float* __restrict__ as_, const float* __restrict__ ad_,
                         const float* __restrict__ mmax, float* __restrict__ denom,
                         float* __restrict__ evals) {
  long long idx = (long long)blockIdx.x * blockDim.x + threadIdx.x;
  if (idx >= ET * NHEAD) return;
  const long long e = idx >> 2; const int hd = (int)(idx & 3);
  long long s, d;
  if (e < E) { s = ei[e]; d = ei[E + e]; } else { s = d = e - E; }
  const float sc = lrelu(as_[s * NHEAD + hd] + ad_[d * NHEAD + hd], 0.2f);
  const float ev = expf(sc - mmax[d * NHEAD + hd]);
  evals[idx] = ev;
  atomicAdd(&denom[d * NHEAD + hd], ev);
}

// agg[dst, h, c] += h[src, h, c] * alpha ; one thread per (edge, h*c)
__global__ void edge_agg(const long long* __restrict__ ei, long long E, long long ET,
                         const float* __restrict__ evals, const float* __restrict__ denom,
                         const float* __restrict__ h, float* __restrict__ agg) {
  long long idx = (long long)blockIdx.x * blockDim.x + threadIdx.x;
  if (idx >= ET * FDIM) return;
  const long long e = idx >> 9;
  const int i = (int)(idx & (FDIM - 1));
  const int hd = i >> 7;
  long long s, d;
  if (e < E) { s = ei[e]; d = ei[E + e]; } else { s = d = e - E; }
  const float alpha = evals[e * NHEAD + hd] / (denom[d * NHEAD + hd] + 1e-16f);
  atomicAdd(&agg[d * FDIM + i], h[s * FDIM + i] * alpha);
}

__global__ void bias_act(float* __restrict__ buf, const float* __restrict__ bias, long long n) {
  long long i = (long long)blockIdx.x * blockDim.x + threadIdx.x;
  if (i >= n) return;
  const int c = (int)(i & (FDIM - 1));
  buf[i] = lrelu(buf[i] + bias[c], 0.01f);
}

__global__ void head_mean(const float* __restrict__ agg, const float* __restrict__ b3,
                          float* __restrict__ out, long long n) {
  long long i = (long long)blockIdx.x * blockDim.x + threadIdx.x;
  if (i >= n) return;
  const long long nd = i >> 7; const int c = (int)(i & (CDIM - 1));
  const float* ap = agg + nd * FDIM;
  out[i] = 0.25f * (ap[c] + ap[CDIM + c] + ap[2 * CDIM + c] + ap[3 * CDIM + c]) + b3[c];
}

__global__ void final_split(const float* __restrict__ z, const float* __restrict__ fcb,
                            float* __restrict__ out, long long nmu) {
  long long i = (long long)blockIdx.x * blockDim.x + threadIdx.x;
  if (i >= nmu) return;
  const long long nd = i >> 6; const int c = (int)(i & 63);
  const float* zr = z + nd * 128;
  out[i]       = zr[c] + fcb[c];                    // mu
  out[nmu + i] = expf(zr[64 + c] + fcb[64 + c]);    // exp(logvar)
}

// ---------------------------------------------------------------------------
extern "C" void kernel_launch(void* const* d_in, const int* in_sizes, int n_in,
                              void* d_out, int out_size, void* d_ws, size_t ws_size,
                              hipStream_t stream) {
  (void)n_in; (void)out_size; (void)ws_size;
  const float*     x   = (const float*)d_in[0];
  const long long* ei  = (const long long*)d_in[1];
  const float* W[3]   = {(const float*)d_in[2], (const float*)d_in[6],  (const float*)d_in[10]};
  const float* atS[3] = {(const float*)d_in[3], (const float*)d_in[7],  (const float*)d_in[11]};
  const float* atD[3] = {(const float*)d_in[4], (const float*)d_in[8],  (const float*)d_in[12]};
  const float* bia[3] = {(const float*)d_in[5], (const float*)d_in[9],  (const float*)d_in[13]};
  const float* fcW = (const float*)d_in[14];
  const float* fcb = (const float*)d_in[15];

  const int       N  = in_sizes[0] / 256;   // 20000 nodes
  const long long E  = in_sizes[1] / 2;     // 320000 edges
  const long long ET = E + N;               // + self loops

  // ---- workspace carve-up (256B-aligned slabs) ----
  char* wsp = (char*)d_ws;
  auto alloc = [&](size_t bytes) {
    char* p = wsp; wsp += (bytes + 255) & ~(size_t)255; return p;
  };
  float*  bufA  = (float*)alloc((size_t)N * FDIM * 4);   // h = X@W       [N,512]
  float*  bufB  = (float*)alloc((size_t)N * FDIM * 4);   // features/agg  [N,512]
  float*  evals = (float*)alloc((size_t)ET * NHEAD * 4); // exp scores    [ET,4]
  float*  a_s   = (float*)alloc((size_t)N * NHEAD * 4);
  float*  a_d   = (float*)alloc((size_t)N * NHEAD * 4);
  float*  mmax  = (float*)alloc((size_t)N * NHEAD * 4);
  float*  denom = (float*)alloc((size_t)N * NHEAD * 4);
  __bf16* Abf   = (__bf16*)alloc((size_t)N * FDIM * 2);  // bf16 activations
  __bf16* Wt    = (__bf16*)alloc((size_t)FDIM * FDIM * 2); // bf16 W^T

  const int mtiles = N / 16;
  dim3 blk(256);
  dim3 ggemm(FDIM / 64, (mtiles + 7) / 8);

  const long long nh    = (long long)N * NHEAD;
  const long long neh   = ET * NHEAD;
  const long long nfeat = (long long)N * FDIM;
  const long long nagg  = ET * FDIM;

  for (int L = 0; L < 3; ++L) {
    const float* in = (L == 0) ? x : bufB;
    const int Kin   = (L == 0) ? 256 : FDIM;
    const long long nin = (long long)N * Kin;
    cvt_bf16<<<(unsigned)((nin + 255) / 256), blk, 0, stream>>>(in, Abf, nin);
    cvt_tr_bf16<<<(Kin * FDIM + 255) / 256, blk, 0, stream>>>(W[L], Wt, Kin, FDIM);
    gemm_wmma_bf16<<<ggemm, blk, 0, stream>>>(Abf, Wt, bufA, N, FDIM, Kin);
    attn_coeff<<<(unsigned)((nh + 255) / 256), blk, 0, stream>>>(bufA, atS[L], atD[L], a_s, a_d, (int)nh);
    fill_f32<<<(unsigned)((nh + 255) / 256), blk, 0, stream>>>(mmax, -1e30f, nh);
    fill_f32<<<(unsigned)((nh + 255) / 256), blk, 0, stream>>>(denom, 0.f, nh);
    fill_f32<<<(unsigned)((nfeat + 255) / 256), blk, 0, stream>>>(bufB, 0.f, nfeat);
    edge_max<<<(unsigned)((neh + 255) / 256), blk, 0, stream>>>(ei, E, ET, a_s, a_d, mmax);
    edge_exp<<<(unsigned)((neh + 255) / 256), blk, 0, stream>>>(ei, E, ET, a_s, a_d, mmax, denom, evals);
    edge_agg<<<(unsigned)((nagg + 255) / 256), blk, 0, stream>>>(ei, E, ET, evals, denom, bufA, bufB);
    if (L < 2)
      bias_act<<<(unsigned)((nfeat + 255) / 256), blk, 0, stream>>>(bufB, bia[L], nfeat);
  }

  // head mean + b3 -> hmean [N,128] (reuse bufA), then FC -> z [N,128]
  float* hmean = bufA;
  float* z     = bufA + (size_t)N * CDIM;
  const long long nmean = (long long)N * CDIM;
  head_mean<<<(unsigned)((nmean + 255) / 256), blk, 0, stream>>>(bufB, bia[2], hmean, nmean);
  cvt_bf16<<<(unsigned)((nmean + 255) / 256), blk, 0, stream>>>(hmean, Abf, nmean);
  cvt_tr_bf16<<<(CDIM * CDIM + 255) / 256, blk, 0, stream>>>(fcW, Wt, CDIM, CDIM);
  dim3 gfc(CDIM / 64, (mtiles + 7) / 8);
  gemm_wmma_bf16<<<gfc, blk, 0, stream>>>(Abf, Wt, z, N, CDIM, CDIM);
  const long long nmu = (long long)N * 64;
  final_split<<<(unsigned)((nmu + 255) / 256), blk, 0, stream>>>(z, fcb, (float*)d_out, nmu);
}
